// CharacterAsWordEmbedding_37288906063902
// MI455X (gfx1250) — matrix operand order repaired and checked by reading, not compiled, not run on hardware
//
#include <hip/hip_runtime.h>
#include <hip/hip_bf16.h>

typedef __attribute__((ext_vector_type(16))) _Float16 v16h;
typedef __attribute__((ext_vector_type(8)))  float    v8f;

#define Bn 128
#define Wn 128
#define Cn 32
#define En 20
#define Hn 64

union AFrag { v16h v; uint4 u[2]; };

// Build a 16x16x32-f16 A fragment from an LDS row (row-major, element = one K).
// Contiguous K order, matching B: lane holds row M = lane&15; halves 0..7 = K[koff..koff+7],
// halves 8..15 = K[koff+8..koff+15], koff = 16*(lane>=16), both 16B-aligned.
__device__ __forceinline__ v16h load_a_lds(const _Float16* row, int koff) {
  AFrag f;
  f.u[0] = *(const uint4*)(row + koff);
  f.u[1] = *(const uint4*)(row + koff + 8);
  return f.v;
}

// Branch-free activations on raw hardware transcendentals (v_exp_f32/v_rcp_f32).
#define LOG2E_F 1.4426950408889634f
__device__ __forceinline__ float sigmoid_(float x) {
  return __builtin_amdgcn_rcpf(1.0f + __builtin_amdgcn_exp2f(-LOG2E_F * x));
}
__device__ __forceinline__ float tanh_(float x) {
  // tanh(x) = 2*sigmoid(2x) - 1
  float s = __builtin_amdgcn_rcpf(1.0f + __builtin_amdgcn_exp2f(-2.0f * LOG2E_F * x));
  return __builtin_fmaf(2.0f, s, -1.0f);
}

__launch_bounds__(128)
__global__ void char_bilstm_kernel(
    const int* __restrict__ chars,
    const float* __restrict__ emb,
    const float* __restrict__ wih_f, const float* __restrict__ whh_f,
    const float* __restrict__ bih_f, const float* __restrict__ bhh_f,
    const float* __restrict__ wih_b,
    const float* __restrict__ bih_b, const float* __restrict__ bhh_b,
    float* __restrict__ out)
{
  // LDS: staged char embeddings (f16, K padded 20->32) + h exchange buffer
  __shared__ __align__(16) _Float16 xtile[Cn * 16 * 32]; // [t][word][k]   32KB
  __shared__ __align__(16) _Float16 hbuf[16 * Hn];       // [word][h]       2KB
  __shared__ int lens_s[16];

  const int tid   = threadIdx.x;
  const int wave  = tid >> 5;        // 0..3  -> hidden cols 16w..16w+15
  const int lane  = tid & 31;
  const int nloc  = lane & 15;       // N (column) within a 16x16 tile
  const int hi    = lane >> 4;       // which half of lanes
  const int koff  = hi ? 16 : 0;     // A-fragment K offset per lane half
  const int arow  = lane & 15;       // A-fragment row (word in tile)
  const int word0 = blockIdx.x * 16;

  // ---- stage char embeddings for the 16-word tile ----
  for (int p = tid; p < Cn * 16; p += 128) {
    int t = p >> 4, m = p & 15;
    int ch = chars[(word0 + m) * Cn + t];
    const float* er = emb + ch * En;
    _Float16* dst = xtile + p * 32;
#pragma unroll
    for (int e = 0; e < En; ++e) dst[e] = (_Float16)er[e];
#pragma unroll
    for (int e = En; e < 32; ++e) dst[e] = (_Float16)0.f;
  }
  if (tid < 16) {
    int cnt = 0;
    for (int t = 0; t < Cn; ++t) cnt += (chars[(word0 + tid) * Cn + t] != 0);
    lens_s[tid] = cnt;
  }
  { // zero the h exchange buffer (128 threads x 8 halves = 1024)
    _Float16* hz = hbuf + tid * 8;
#pragma unroll
    for (int j = 0; j < 8; ++j) hz[j] = (_Float16)0.f;
  }
  __syncthreads();

  // per-lane (word row) capture info for the 8 accumulator rows
  int   idxv[8];
  float validv[8];
#pragma unroll
  for (int v = 0; v < 8; ++v) {
    int len   = lens_s[hi * 8 + v];
    idxv[v]   = (len > 0) ? (len - 1) : 0;
    validv[v] = (len > 0) ? 1.0f : 0.0f;
  }

  // ---- weight fragments, resident in VGPRs for the whole recurrence ----
  // B-fragment layout: lane's column n fixed; halves j=0..15 = K(kbase+j),
  // kbase = 16*(lane>=16). => 16 contiguous K values of row n.
  const int kbase = hi * 16;
  v16h  bih[4];      // input weights  (K=0..31, 20..31 zero-padded)
  v16h  bhh[4][2];   // recurrent weights, K chunks [0..31], [32..63]
  float biasf[4];
#pragma unroll
  for (int G = 0; G < 4; ++G) {
    int n = G * 64 + wave * 16 + nloc;           // torch gate order i,f,g,o
    v16h f;
#pragma unroll
    for (int j = 0; j < 16; ++j) {
      int k = kbase + j;
      f[j] = (k < En) ? (_Float16)wih_f[n * En + k] : (_Float16)0.f;
    }
    bih[G] = f;
#pragma unroll
    for (int q = 0; q < 2; ++q) {
      v16h g;
#pragma unroll
      for (int j = 0; j < 16; ++j)
        g[j] = (_Float16)whh_f[n * Hn + q * 32 + kbase + j];
      bhh[G][q] = g;
    }
    biasf[G] = bih_f[n] + bhh_f[n];
  }

  float c_st[8], hf[8];
#pragma unroll
  for (int v = 0; v < 8; ++v) { c_st[v] = 0.f; hf[v] = 0.f; }

  const _Float16* hptr = hbuf + arow * Hn;

  // ---- 32-step forward recurrence ----
  for (int t = 0; t < Cn; ++t) {
    v16h a_x  = load_a_lds(xtile + (t * 16 + arow) * 32, koff);
    v16h a_h0 = load_a_lds(hptr, koff);
    v16h a_h1 = load_a_lds(hptr, 32 + koff);

    v8f acc[4];
#pragma unroll
    for (int G = 0; G < 4; ++G) {
      v8f a;
#pragma unroll
      for (int j = 0; j < 8; ++j) a[j] = biasf[G];
      a = __builtin_amdgcn_wmma_f32_16x16x32_f16(false, a_x,  false, bih[G],
                                                 (short)0, a, false, false);
      a = __builtin_amdgcn_wmma_f32_16x16x32_f16(false, a_h0, false, bhh[G][0],
                                                 (short)0, a, false, false);
      a = __builtin_amdgcn_wmma_f32_16x16x32_f16(false, a_h1, false, bhh[G][1],
                                                 (short)0, a, false, false);
      acc[G] = a;
    }

    // lane-local LSTM cell update: acc element v = (word hi*8+v, hidden 16w+nloc)
    float hnew[8];
#pragma unroll
    for (int v = 0; v < 8; ++v) {
      float ig = sigmoid_(acc[0][v]);
      float fg = sigmoid_(acc[1][v]);
      float gg = tanh_(acc[2][v]);
      float og = sigmoid_(acc[3][v]);
      float c  = fg * c_st[v] + ig * gg;
      c_st[v]  = c;
      float h  = og * tanh_(c);
      hnew[v]  = h;
      hf[v]    = (t == idxv[v]) ? h : hf[v];   // capture at last valid char
    }

    __syncthreads();                           // all reads of hbuf done
#pragma unroll
    for (int v = 0; v < 8; ++v)
      hbuf[(hi * 8 + v) * Hn + wave * 16 + nloc] = (_Float16)hnew[v];
    __syncthreads();                           // new h visible
  }

  // ---- forward outputs ----
#pragma unroll
  for (int v = 0; v < 8; ++v) {
    int wglob = word0 + hi * 8 + v;
    out[wglob * (2 * Hn) + wave * 16 + nloc] = hf[v] * validv[v];
  }

  // ---- backward direction: one LSTM step on last char from zero state ----
  v16h  bihb[4];
  float biasb[4];
#pragma unroll
  for (int G = 0; G < 4; ++G) {
    int n = G * 64 + wave * 16 + nloc;
    v16h f;
#pragma unroll
    for (int j = 0; j < 16; ++j) {
      int k = kbase + j;
      f[j] = (k < En) ? (_Float16)wih_b[n * En + k] : (_Float16)0.f;
    }
    bihb[G] = f;
    biasb[G] = bih_b[n] + bhh_b[n];
  }
  int tlast = (lens_s[arow] > 0) ? lens_s[arow] - 1 : 0;  // per-lane row's step
  v16h a_xl = load_a_lds(xtile + (tlast * 16 + arow) * 32, koff);

  v8f accb[4];
#pragma unroll
  for (int G = 0; G < 4; ++G) {
    v8f a;
#pragma unroll
    for (int j = 0; j < 8; ++j) a[j] = biasb[G];
    accb[G] = __builtin_amdgcn_wmma_f32_16x16x32_f16(false, a_xl, false, bihb[G],
                                                     (short)0, a, false, false);
  }
#pragma unroll
  for (int v = 0; v < 8; ++v) {
    float cb = sigmoid_(accb[0][v]) * tanh_(accb[2][v]);  // f-gate * 0 drops out
    float hb = sigmoid_(accb[3][v]) * tanh_(cb);
    int wglob = word0 + hi * 8 + v;
    out[wglob * (2 * Hn) + Hn + wave * 16 + nloc] = hb * validv[v];
  }
}

extern "C" void kernel_launch(void* const* d_in, const int* in_sizes, int n_in,
                              void* d_out, int out_size, void* d_ws, size_t ws_size,
                              hipStream_t stream) {
  const int*   chars = (const int*)d_in[0];
  const float* emb   = (const float*)d_in[1];
  const float* wih_f = (const float*)d_in[2];
  const float* whh_f = (const float*)d_in[3];
  const float* bih_f = (const float*)d_in[4];
  const float* bhh_f = (const float*)d_in[5];
  const float* wih_b = (const float*)d_in[6];
  // d_in[7] (w_hh_b) multiplies the zero initial state -> unused
  const float* bih_b = (const float*)d_in[8];
  const float* bhh_b = (const float*)d_in[9];
  float* out = (float*)d_out;

  dim3 grid((Bn * Wn) / 16);   // 1024 tiles of 16 words
  dim3 blk(128);               // 4 waves (wave32)
  hipLaunchKernelGGL(char_bilstm_kernel, grid, blk, 0, stream,
                     chars, emb, wih_f, whh_f, bih_f, bhh_f,
                     wih_b, bih_b, bhh_b, out);
}
